// rnn_pack_encoder_47124381172130
// MI455X (gfx1250) — matrix operand order repaired and checked by reading, not compiled, not run on hardware
//
#include <hip/hip_runtime.h>
#include <hip/hip_bf16.h>
#include <stdint.h>

#define BB 32
#define TT 1024
#define DD 1024
#define HH 1024
#define G3 3072
#define NBLK 128
#define LDSPAD 40   // bf16 elements per staged B row (32 data + 8 pad)

typedef __attribute__((ext_vector_type(16))) __bf16 v16bf;
typedef __attribute__((ext_vector_type(8)))  __bf16 v8bf;
typedef __attribute__((ext_vector_type(8)))  float  v8f;

union V16 { v16bf v; v8bf h[2]; };

__device__ __forceinline__ unsigned short f2bf(float f) {
  unsigned u = __float_as_uint(f);
  return (unsigned short)((u + 0x7FFFu + ((u >> 16) & 1u)) >> 16);  // RNE
}

// async copy of 16 bytes global -> LDS, per lane (ASYNCcnt-tracked)
__device__ __forceinline__ void async_b128_to_lds(unsigned lds_off,
                                                  const void* gaddr) {
  asm volatile("global_load_async_to_lds_b128 %0, %1, off"
               :: "v"(lds_off), "v"(gaddr) : "memory");
}

// ---------------------------------------------------------------- utilities
__global__ void zero_u32_kernel(unsigned* __restrict__ p, long long n) {
  long long i = (long long)blockIdx.x * blockDim.x + threadIdx.x;
  long long s = (long long)gridDim.x * blockDim.x;
  for (; i < n; i += s) p[i] = 0u;
}

__global__ void f32_to_bf16_kernel(const float* __restrict__ in,
                                   unsigned short* __restrict__ out, int n) {
  int i = blockIdx.x * blockDim.x + threadIdx.x;
  int s = gridDim.x * blockDim.x;
  for (; i < n; i += s) out[i] = f2bf(in[i]);
}

// emission positions (exclusive cumsum of seg==0) and counts
__global__ void seg_scan_kernel(const int* __restrict__ seg,
                                int* __restrict__ pos,
                                int* __restrict__ counts_out) {
  int b = threadIdx.x;
  if (b >= BB) return;
  int cnt = 0;
  for (int t = 0; t < TT; ++t) {
    int s = seg[b * TT + t];
    if (s == 0) { pos[b * TT + t] = cnt; ++cnt; }
    else        { pos[b * TT + t] = TT; }
  }
  counts_out[b] = cnt;
}

// ---------------------------------------------------------------- gi GEMM
// gi[r, :] = x[r, :] @ W_ih^T + b_ih   (r = b*T + t flattened)
// block = 128 threads (4 waves), tile 64(M) x 64(N).
// The B tile (shared by all 4 waves) is staged to LDS with async copies,
// double-buffered; A fragments are per-wave direct global_load_b128.
__launch_bounds__(128, 1)
__global__ void gi_gemm_kernel(const unsigned short* __restrict__ xbf,
                               const unsigned short* __restrict__ wbf,
                               const float* __restrict__ bias,
                               float* __restrict__ gi) {
  __shared__ unsigned short bsm[2][64 * LDSPAD];   // 2 x 5120 B

  const int tid  = threadIdx.x;
  const int lane = tid & 31;
  const int w    = tid >> 5;
  const int lm   = lane & 15;
  const int lh   = lane >> 4;
  const int mrow = blockIdx.y * 64 + w * 16 + lm;   // A row this lane loads
  const int n0   = blockIdx.x * 64;

  // this thread's two 16-byte staging chunks of the 64x32 B tile
  const int c0   = tid * 2;
  const int row0 = (c0)     >> 2, q0 = (c0)     & 3;
  const int row1 = (c0 + 1) >> 2, q1 = (c0 + 1) & 3;
  const unsigned short* g0 = wbf + (size_t)(n0 + row0) * DD + q0 * 8;
  const unsigned short* g1 = wbf + (size_t)(n0 + row1) * DD + q1 * 8;
  const unsigned l0 = (unsigned)(uintptr_t)&bsm[0][row0 * LDSPAD + q0 * 8];
  const unsigned l1 = (unsigned)(uintptr_t)&bsm[0][row1 * LDSPAD + q1 * 8];
  const unsigned lstep = (unsigned)(uintptr_t)&bsm[1][0] -
                         (unsigned)(uintptr_t)&bsm[0][0];

  v8f acc[4] = {};
  const unsigned short* ap = xbf + (size_t)mrow * DD;

  // prologue: stage k0 = 0 into buffer 0
  async_b128_to_lds(l0, g0);
  async_b128_to_lds(l1, g1);

  int kb = 0;
  for (int k0 = 0; k0 < DD; k0 += 32, kb ^= 1) {
    if (k0 + 32 < DD) {
      // stage next K-slice into the other buffer
      async_b128_to_lds(l0 + (kb ^ 1) * lstep, g0 + k0 + 32);
      async_b128_to_lds(l1 + (kb ^ 1) * lstep, g1 + k0 + 32);
      asm volatile("s_wait_asynccnt 2" ::: "memory");
    } else {
      asm volatile("s_wait_asynccnt 0" ::: "memory");
    }
    __syncthreads();   // whole B tile for k0 is resident

    V16 a;
    const unsigned short* a0 = ap + k0 + lh * 16;    // lanes 0-15: K 0-15
    a.h[0] = *reinterpret_cast<const v8bf*>(a0);     // lanes16-31: K 16-31
    a.h[1] = *reinterpret_cast<const v8bf*>(a0 + 8);
#pragma unroll
    for (int j = 0; j < 4; ++j) {
      const unsigned short* bp = &bsm[kb][(j * 16 + lm) * LDSPAD + lh * 16];
      V16 bm;
      bm.h[0] = *reinterpret_cast<const v8bf*>(bp);
      bm.h[1] = *reinterpret_cast<const v8bf*>(bp + 8);
      acc[j] = __builtin_amdgcn_wmma_f32_16x16x32_bf16(
          false, a.v, false, bm.v, (short)0, acc[j], false, false);
    }
    __syncthreads();   // protect bsm[kb] before it is re-staged at k0+64
  }

  const int mbase = blockIdx.y * 64 + w * 16 + lh * 8;  // C layout: M = r + lh*8
#pragma unroll
  for (int j = 0; j < 4; ++j) {
    const int col = n0 + j * 16 + lm;                   // C layout: N = lane%16
    const float bv = bias[col];
#pragma unroll
    for (int r = 0; r < 8; ++r)
      gi[(size_t)(mbase + r) * G3 + col] = acc[j][r] + bv;
  }
}

// ---------------------------------------------------------------- recurrent scan
// 128 persistent blocks: unit = (batch row tile 0..1) x (hidden col tile 0..63).
// 4 waves split K=1024; LDS reduce; wave 0 applies gates, emits, updates h.
__launch_bounds__(128, 1)
__global__ void gru_scan_kernel(const unsigned short* __restrict__ whhbf,
                                const float* __restrict__ b_hh,
                                const float* __restrict__ gi,
                                const int* __restrict__ seg,
                                const int* __restrict__ pos,
                                unsigned short* __restrict__ hbf, // 2 * BB*HH bf16
                                float* __restrict__ hf,           // 2 * BB*HH f32
                                float* __restrict__ out,
                                unsigned* __restrict__ sync_counter,
                                unsigned* __restrict__ sync_gen) {
  __shared__ float red[4][3][8][32];
  const int lane = threadIdx.x & 31;
  const int w    = threadIdx.x >> 5;
  const int lm   = lane & 15;
  const int lh   = lane >> 4;
  const int unit = blockIdx.x;
  const int rt   = unit >> 6;   // batch row tile (0..1)
  const int ct   = unit & 63;   // hidden col tile (0..63)
  const int kbase = w * 256;    // this wave's K slice

  for (int t = 0; t < TT; ++t) {
    const unsigned short* hbc = hbf + ((t)     & 1) * (BB * HH);
    unsigned short*       hbn = hbf + ((t + 1) & 1) * (BB * HH);
    const float*          hfc = hf  + ((t)     & 1) * (BB * HH);
    float*                hfn = hf  + ((t + 1) & 1) * (BB * HH);

    v8f acc[3] = {};
    const int arow = rt * 16 + lm;
    const unsigned short* ap = hbc + arow * HH;

    for (int k0 = kbase; k0 < kbase + 256; k0 += 32) {
      V16 a;
      const unsigned short* a0 = ap + k0 + lh * 16;
      a.h[0] = *reinterpret_cast<const v8bf*>(a0);
      a.h[1] = *reinterpret_cast<const v8bf*>(a0 + 8);
#pragma unroll
      for (int g = 0; g < 3; ++g) {
        const int orow = g * HH + ct * 16 + lm;       // W_hh row
        const unsigned short* bp = whhbf + (size_t)orow * HH + k0 + lh * 16;
        V16 bm;
        bm.h[0] = *reinterpret_cast<const v8bf*>(bp);
        bm.h[1] = *reinterpret_cast<const v8bf*>(bp + 8);
        acc[g] = __builtin_amdgcn_wmma_f32_16x16x32_bf16(
            false, a.v, false, bm.v, (short)0, acc[g], false, false);
      }
    }

#pragma unroll
    for (int g = 0; g < 3; ++g)
#pragma unroll
      for (int r = 0; r < 8; ++r)
        red[w][g][r][lane] = acc[g][r];
    __syncthreads();

    if (w == 0) {
      const int col = ct * 16 + lm;
      const float bR = b_hh[col], bZ = b_hh[HH + col], bN = b_hh[2 * HH + col];
#pragma unroll
      for (int r = 0; r < 8; ++r) {
        float sR = red[0][0][r][lane] + red[1][0][r][lane] +
                   red[2][0][r][lane] + red[3][0][r][lane];
        float sZ = red[0][1][r][lane] + red[1][1][r][lane] +
                   red[2][1][r][lane] + red[3][1][r][lane];
        float sN = red[0][2][r][lane] + red[1][2][r][lane] +
                   red[2][2][r][lane] + red[3][2][r][lane];
        const int brow = rt * 16 + lh * 8 + r;
        const float* gip = gi + ((size_t)brow * TT + t) * G3;
        float ir  = gip[col];
        float iz  = gip[HH + col];
        float inn = gip[2 * HH + col];
        float rg = 1.f / (1.f + __expf(-(ir + sR + bR)));
        float zg = 1.f / (1.f + __expf(-(iz + sZ + bZ)));
        float nx = inn + rg * (sN + bN);
        float e  = __expf(-2.f * nx);
        float ng = (1.f - e) / (1.f + e);
        float hprev = hfc[brow * HH + col];
        float hnew  = (1.f - zg) * ng + zg * hprev;
        int s = seg[brow * TT + t];
        if (s == 0) {
          int p = pos[brow * TT + t];
          out[((size_t)brow * TT + p) * HH + col] = hnew;
        }
        float hx = s ? hnew : 0.f;
        hfn[brow * HH + col] = hx;
        hbn[brow * HH + col] = f2bf(hx);
      }
    } else if (t + 1 < TT) {
      // idle waves prefetch next step's gi slices for this unit (48 lines)
      int idx = (w - 1) * 32 + lane;
      if (idx < 48) {
        int brow = rt * 16 + (idx & 15);
        int g = idx >> 4;
        __builtin_prefetch(gi + ((size_t)brow * TT + (t + 1)) * G3 +
                               g * HH + ct * 16, 0, 3);
      }
    }

    // device-wide generation barrier between timesteps
    __syncthreads();
    if (threadIdx.x == 0) {
      __threadfence();
      unsigned g = __hip_atomic_load(sync_gen, __ATOMIC_ACQUIRE,
                                     __HIP_MEMORY_SCOPE_AGENT);
      unsigned a = __hip_atomic_fetch_add(sync_counter, 1u, __ATOMIC_ACQ_REL,
                                          __HIP_MEMORY_SCOPE_AGENT);
      if (a == NBLK - 1) {
        __hip_atomic_store(sync_counter, 0u, __ATOMIC_RELAXED,
                           __HIP_MEMORY_SCOPE_AGENT);
        __hip_atomic_store(sync_gen, g + 1u, __ATOMIC_RELEASE,
                           __HIP_MEMORY_SCOPE_AGENT);
      } else {
        while (__hip_atomic_load(sync_gen, __ATOMIC_ACQUIRE,
                                 __HIP_MEMORY_SCOPE_AGENT) == g) {
          __builtin_amdgcn_s_sleep(2);
        }
      }
    }
    __syncthreads();
  }
}

// ---------------------------------------------------------------- launcher
extern "C" void kernel_launch(void* const* d_in, const int* in_sizes, int n_in,
                              void* d_out, int out_size, void* d_ws, size_t ws_size,
                              hipStream_t stream) {
  (void)in_sizes; (void)n_in; (void)out_size; (void)ws_size;
  const float* x   = (const float*)d_in[0];
  const float* Wih = (const float*)d_in[1];
  const float* Whh = (const float*)d_in[2];
  const float* bih = (const float*)d_in[3];
  const float* bhh = (const float*)d_in[4];
  const int*   seg = (const int*)d_in[5];
  float* out = (float*)d_out;

  char* ws = (char*)d_ws;
  size_t off = 0;
  auto alloc = [&](size_t bytes) -> void* {
    void* p = ws + off;
    off += (bytes + 255) & ~(size_t)255;
    return p;
  };
  unsigned short* xbf  = (unsigned short*)alloc((size_t)BB * TT * DD * 2);
  unsigned short* wihb = (unsigned short*)alloc((size_t)G3 * DD * 2);
  unsigned short* whhb = (unsigned short*)alloc((size_t)G3 * HH * 2);
  float*          gi   = (float*)alloc((size_t)BB * TT * G3 * 4);
  unsigned short* hbf  = (unsigned short*)alloc((size_t)2 * BB * HH * 2);
  float*          hf   = (float*)alloc((size_t)2 * BB * HH * 4);
  int*            pos  = (int*)alloc((size_t)BB * TT * 4);
  unsigned*       syncb = (unsigned*)alloc(256);

  // init: zero output region, h double-buffers, sync cells
  zero_u32_kernel<<<4096, 256, 0, stream>>>((unsigned*)out, (long long)BB * TT * HH);
  zero_u32_kernel<<<64, 256, 0, stream>>>((unsigned*)hbf, (long long)(BB * HH)); // 2*BB*HH u16
  zero_u32_kernel<<<64, 256, 0, stream>>>((unsigned*)hf, (long long)(2 * BB * HH));
  zero_u32_kernel<<<1, 64, 0, stream>>>(syncb, 64);

  // bf16 conversion
  f32_to_bf16_kernel<<<8192, 256, 0, stream>>>(x, xbf, BB * TT * DD);
  f32_to_bf16_kernel<<<2048, 256, 0, stream>>>(Wih, wihb, G3 * DD);
  f32_to_bf16_kernel<<<2048, 256, 0, stream>>>(Whh, whhb, G3 * HH);

  // emission positions + counts (counts go to tail of d_out as int32)
  seg_scan_kernel<<<1, 32, 0, stream>>>(seg, pos, (int*)(out + (size_t)BB * TT * HH));

  // gi = x @ W_ih^T + b_ih
  dim3 gg(G3 / 64, (BB * TT) / 64);
  gi_gemm_kernel<<<gg, 128, 0, stream>>>(xbf, wihb, bih, gi);

  // persistent recurrent scan
  gru_scan_kernel<<<NBLK, 128, 0, stream>>>(whhb, bhh, gi, seg, pos,
                                            hbf, hf, out, syncb, syncb + 1);
}
